// MultiHeadAttention_26156350832790
// MI455X (gfx1250) — compile-verified, matrix operand order checked by reading, not run on hardware
//
#include <hip/hip_runtime.h>
#include <hip/hip_bf16.h>

// ---------------------------------------------------------------------------
// MHA for MI455X (gfx1250). All matmuls via v_wmma_f32_16x16x32_bf16 with
// fragments loaded directly from global (L2-resident bf16 operand copies),
// so the inner loops are pure {global_load_b128 x16 ; v_wmma x16} streams.
//
// d_ws layout (bf16 elems, 4M each = 8 MB):
//   [0]  x_bf16 [2048][2048]      (row-major)   -- later reused as attn buffer
//   [1]  WqT    [2048][2048]      (N x K, transposed bf16)
//   [2]  WkT    [2048][2048]
//   [3]  WvT    [2048][2048]
//   [4]  WoT    [2048][2048]
//   [5]  q      [16][2048][128]
//   [6]  k      [16][2048][128]
//   [7]  vT     [16][128][2048]
// Total 64 MB.
// ---------------------------------------------------------------------------

typedef __bf16 bf16_t;
typedef __attribute__((ext_vector_type(16))) __bf16 v16bf;
typedef __attribute__((ext_vector_type(8)))  float  v8f;

union AFrag { v16bf v; uint4 q[2]; };
union Pack4 { bf16_t h[4]; uint2 u; };

__device__ __forceinline__ v8f wmma_bf16(v16bf a, v16bf b, v8f c) {
  // 8 args: (neg_a, A, neg_b, B, c_mod, C, reuse_a, reuse_b)
  return __builtin_amdgcn_wmma_f32_16x16x32_bf16(false, a, false, b,
                                                 (short)0, c, false, false);
}

#define D_MODEL 2048
#define HEADS   16
#define DK      128

// ---------------------------------------------------------------------------
// Elementwise f32 -> bf16 convert (x). float4 in, 4x bf16 out.
// ---------------------------------------------------------------------------
__global__ __launch_bounds__(256)
void convert_bf16(const float* __restrict__ src, bf16_t* __restrict__ dst)
{
  const size_t i = (size_t)blockIdx.x * blockDim.x + threadIdx.x;   // float4 idx
  const float4 f = ((const float4*)src)[i];
  Pack4 p;
  p.h[0] = (bf16_t)f.x; p.h[1] = (bf16_t)f.y;
  p.h[2] = (bf16_t)f.z; p.h[3] = (bf16_t)f.w;
  ((uint2*)dst)[i] = p.u;
}

// ---------------------------------------------------------------------------
// Transpose + convert: WT[n][k] = (bf16) W[k][n].  32x32 tiles via padded LDS.
// ---------------------------------------------------------------------------
__global__ __launch_bounds__(256)
void transpose_convert_bf16(const float* __restrict__ W, bf16_t* __restrict__ WT)
{
  __shared__ float T[32][33];
  const int tx = threadIdx.x;            // 0..31
  const int ty = threadIdx.y;            // 0..7
  const int k0 = blockIdx.y * 32;
  const int n0 = blockIdx.x * 32;
#pragma unroll
  for (int i = 0; i < 4; ++i)
    T[ty + 8 * i][tx] = W[(size_t)(k0 + ty + 8 * i) * D_MODEL + (n0 + tx)];
  __syncthreads();
#pragma unroll
  for (int i = 0; i < 4; ++i)
    WT[(size_t)(n0 + ty + 8 * i) * D_MODEL + (k0 + tx)] =
        (bf16_t)T[tx][ty + 8 * i];
}

// ---------------------------------------------------------------------------
// LDS-free GEMM: C[2048x2048] = A(bf16 row-major) * BT(bf16 NxK)^T  (+bias)
// 128 threads = 4 waves; block tile 128x128; wave tile 64x64 (4x4 frags).
// Per 32-k chunk: 8 global b128 load-pairs feed 16 WMMAs, no barriers.
// Fragment layout per ISA (16-bit A/B):
//   lane l<16 : row/col l,   contraction elems {c*32+0..7,  c*32+16..23}
//   lane l>=16: row/col l-16, contraction elems {c*32+8..15, c*32+24..31}
// MODE 0: out bf16, Q/K layout [H][S][DK]
// MODE 1: out bf16, V  layout [H][DK][S]
// MODE 2: out f32 row-major (+bias)
// ---------------------------------------------------------------------------
template <int MODE>
__global__ __launch_bounds__(128)
void gemm_bf16_wmma(const bf16_t* __restrict__ A, const bf16_t* __restrict__ BT,
                    const float* __restrict__ bias, void* __restrict__ outp)
{
  const int tid  = threadIdx.x;
  const int wave = tid >> 5;
  const int lane = tid & 31;
  const int half = lane >> 4;
  const int l16  = lane & 15;
  const int bm = blockIdx.y * 128 + (wave >> 1) * 64;
  const int bn = blockIdx.x * 128 + (wave & 1)  * 64;

  const bf16_t* ap[4];
  const bf16_t* bp[4];
#pragma unroll
  for (int mi = 0; mi < 4; ++mi)
    ap[mi] = A + (size_t)(bm + mi * 16 + l16) * D_MODEL + half * 8;
#pragma unroll
  for (int ni = 0; ni < 4; ++ni)
    bp[ni] = BT + (size_t)(bn + ni * 16 + l16) * D_MODEL + half * 8;

  v8f acc[4][4] = {};

  for (int k0 = 0; k0 < D_MODEL; k0 += 32) {
    AFrag af[4], bf[4];
#pragma unroll
    for (int mi = 0; mi < 4; ++mi) {
      af[mi].q[0] = *(const uint4*)(ap[mi]);
      af[mi].q[1] = *(const uint4*)(ap[mi] + 16);
      ap[mi] += 32;
    }
#pragma unroll
    for (int ni = 0; ni < 4; ++ni) {
      bf[ni].q[0] = *(const uint4*)(bp[ni]);
      bf[ni].q[1] = *(const uint4*)(bp[ni] + 16);
      bp[ni] += 32;
    }
    if (k0 + 32 < D_MODEL) {            // hint next k-chunk toward the caches
      __builtin_prefetch(ap[0], 0, 0);
      __builtin_prefetch(bp[0], 0, 0);
    }
#pragma unroll
    for (int mi = 0; mi < 4; ++mi)
#pragma unroll
      for (int ni = 0; ni < 4; ++ni)
        acc[mi][ni] = wmma_bf16(af[mi].v, bf[ni].v, acc[mi][ni]);
  }

  // C layout: lane l16 = column, VGPR j = row j (+8 for the hi half)
#pragma unroll
  for (int mi = 0; mi < 4; ++mi)
#pragma unroll
    for (int ni = 0; ni < 4; ++ni)
#pragma unroll
      for (int j = 0; j < 8; ++j) {
        const int gm = bm + mi * 16 + j + half * 8;
        const int gn = bn + ni * 16 + l16;
        const float v = acc[mi][ni][j] + bias[gn];
        if (MODE == 0) {
          bf16_t* o = (bf16_t*)outp;
          const int h = gn >> 7, d = gn & 127;
          o[((size_t)h * D_MODEL + gm) * DK + d] = (bf16_t)v;
        } else if (MODE == 1) {
          bf16_t* o = (bf16_t*)outp;
          const int h = gn >> 7, d = gn & 127;
          o[((size_t)h * DK + d) * D_MODEL + gm] = (bf16_t)v;
        } else {
          float* o = (float*)outp;
          o[(size_t)gm * D_MODEL + gn] = v;
        }
      }
}

// ---------------------------------------------------------------------------
// Flash attention, causal. One wave per (head, 16-query tile).
// Per 32-key block: 8 WMMA for QK^T, online softmax (shfl_xor row reductions
// within 16-lane halves), P transposed via LDS to A-fragment layout, 8 WMMA
// for PV against V stored transposed. Output attn[S][D] bf16.
// ---------------------------------------------------------------------------
__global__ __launch_bounds__(32)
void flash_attn_wmma(const bf16_t* __restrict__ q, const bf16_t* __restrict__ k,
                     const bf16_t* __restrict__ vT, bf16_t* __restrict__ attn)
{
  __shared__ bf16_t Psm[16][32];          // [query m][key 0..31]

  const int lane = threadIdx.x;
  const int half = lane >> 4;
  const int l16  = lane & 15;
  const int h    = blockIdx.y;
  const int qb   = blockIdx.x * 16;

  const bf16_t* qh = q  + (size_t)h * D_MODEL * DK;
  const bf16_t* kh = k  + (size_t)h * D_MODEL * DK;
  const bf16_t* vh = vT + (size_t)h * DK * D_MODEL;

  AFrag aq[4];
  {
    const bf16_t* qrow = qh + (size_t)(qb + l16) * DK;
#pragma unroll
    for (int c = 0; c < 4; ++c) {
      const bf16_t* p = qrow + c * 32 + half * 8;
      aq[c].q[0] = *(const uint4*)(p);
      aq[c].q[1] = *(const uint4*)(p + 16);
    }
  }

  v8f acc[8] = {};
  float mrow[8], lrow[8];
#pragma unroll
  for (int j = 0; j < 8; ++j) { mrow[j] = -3.0e38f; lrow[j] = 0.0f; }

  const float scale = 0.08838834764831845f;   // 1/sqrt(128)
  const int nkb = (qb + 47) >> 5;

  for (int kb0 = 0; kb0 < nkb; ++kb0) {
    const int kb = kb0 * 32;

    v8f s0 = {}, s1 = {};
#pragma unroll
    for (int c = 0; c < 4; ++c) {
      AFrag bk0, bk1;
      const bf16_t* kr0 = kh + (size_t)(kb + l16) * DK      + c * 32 + half * 8;
      const bf16_t* kr1 = kh + (size_t)(kb + 16 + l16) * DK + c * 32 + half * 8;
      bk0.q[0] = *(const uint4*)(kr0); bk0.q[1] = *(const uint4*)(kr0 + 16);
      bk1.q[0] = *(const uint4*)(kr1); bk1.q[1] = *(const uint4*)(kr1 + 16);
      s0 = wmma_bf16(aq[c].v, bk0.v, s0);
      s1 = wmma_bf16(aq[c].v, bk1.v, s1);
    }

    float p0[8], p1[8];
#pragma unroll
    for (int j = 0; j < 8; ++j) {
      const int gm = qb + j + half * 8;
      float x0 = s0[j] * scale; if (kb + l16      > gm) x0 = -3.0e38f;
      float x1 = s1[j] * scale; if (kb + 16 + l16 > gm) x1 = -3.0e38f;
      p0[j] = x0; p1[j] = x1;
    }
#pragma unroll
    for (int j = 0; j < 8; ++j) {
      float mx = fmaxf(p0[j], p1[j]);
#pragma unroll
      for (int off = 1; off < 16; off <<= 1)
        mx = fmaxf(mx, __shfl_xor(mx, off, 32));
      const float mnew  = fmaxf(mrow[j], mx);
      const float alpha = __expf(mrow[j] - mnew);
      const float e0 = __expf(p0[j] - mnew);
      const float e1 = __expf(p1[j] - mnew);
      float rs = e0 + e1;
#pragma unroll
      for (int off = 1; off < 16; off <<= 1)
        rs += __shfl_xor(rs, off, 32);
      lrow[j] = lrow[j] * alpha + rs;
      mrow[j] = mnew;
      p0[j] = e0; p1[j] = e1;
#pragma unroll
      for (int t = 0; t < 8; ++t) acc[t][j] *= alpha;
    }

    __syncthreads();
#pragma unroll
    for (int j = 0; j < 8; ++j) {
      Psm[j + half * 8][l16]      = (bf16_t)p0[j];
      Psm[j + half * 8][16 + l16] = (bf16_t)p1[j];
    }
    __syncthreads();
    AFrag pa;
    {
      const bf16_t* pr = &Psm[l16][half * 8];
      pa.q[0] = *(const uint4*)(pr);
      pa.q[1] = *(const uint4*)(pr + 16);
    }

#pragma unroll
    for (int t = 0; t < 8; ++t) {
      AFrag bv;
      const bf16_t* vr = vh + (size_t)(t * 16 + l16) * D_MODEL + kb + half * 8;
      bv.q[0] = *(const uint4*)(vr);
      bv.q[1] = *(const uint4*)(vr + 16);
      acc[t] = wmma_bf16(pa.v, bv.v, acc[t]);
    }
  }

  float inv[8];
#pragma unroll
  for (int j = 0; j < 8; ++j) inv[j] = 1.0f / lrow[j];
#pragma unroll
  for (int t = 0; t < 8; ++t)
#pragma unroll
    for (int j = 0; j < 8; ++j) {
      const int gm = qb + j + half * 8;
      const int gn = h * DK + t * 16 + l16;
      attn[(size_t)gm * D_MODEL + gn] = (bf16_t)(acc[t][j] * inv[j]);
    }
}

// ---------------------------------------------------------------------------
extern "C" void kernel_launch(void* const* d_in, const int* in_sizes, int n_in,
                              void* d_out, int out_size, void* d_ws, size_t ws_size,
                              hipStream_t stream) {
  (void)in_sizes; (void)n_in; (void)out_size; (void)ws_size;
  const float* x  = (const float*)d_in[0];
  const float* Wq = (const float*)d_in[1];
  const float* bq = (const float*)d_in[2];
  const float* Wk = (const float*)d_in[3];
  const float* bk = (const float*)d_in[4];
  const float* Wv = (const float*)d_in[5];
  const float* bv = (const float*)d_in[6];
  const float* Wo = (const float*)d_in[7];
  const float* bo = (const float*)d_in[8];

  const size_t elems = (size_t)D_MODEL * D_MODEL;   // 4M bf16 = 8 MB
  bf16_t* xb   = (bf16_t*)d_ws;
  bf16_t* wqT  = xb  + elems;
  bf16_t* wkT  = wqT + elems;
  bf16_t* wvT  = wkT + elems;
  bf16_t* woT  = wvT + elems;
  bf16_t* qd   = woT + elems;
  bf16_t* kd   = qd  + elems;
  bf16_t* vTd  = kd  + elems;
  bf16_t* attn = xb;   // x_bf16 is dead once Q/K/V are built; reuse for attn

  // 1) operand packing: x -> bf16 row-major; W* -> bf16 transposed (N x K)
  convert_bf16<<<dim3(elems / 4 / 256), dim3(256), 0, stream>>>(x, xb);
  dim3 tgrid(D_MODEL / 32, D_MODEL / 32), tblk(32, 8);
  transpose_convert_bf16<<<tgrid, tblk, 0, stream>>>(Wq, wqT);
  transpose_convert_bf16<<<tgrid, tblk, 0, stream>>>(Wk, wkT);
  transpose_convert_bf16<<<tgrid, tblk, 0, stream>>>(Wv, wvT);
  transpose_convert_bf16<<<tgrid, tblk, 0, stream>>>(Wo, woT);

  // 2) QKV projections (LDS-free WMMA GEMMs)
  dim3 ggrid(D_MODEL / 128, D_MODEL / 128), gblk(128);
  gemm_bf16_wmma<0><<<ggrid, gblk, 0, stream>>>(xb, wqT, bq, (void*)qd);
  gemm_bf16_wmma<0><<<ggrid, gblk, 0, stream>>>(xb, wkT, bk, (void*)kd);
  gemm_bf16_wmma<1><<<ggrid, gblk, 0, stream>>>(xb, wvT, bv, (void*)vTd);

  // 3) causal flash attention
  dim3 agrid(D_MODEL / 16, HEADS);
  flash_attn_wmma<<<agrid, dim3(32), 0, stream>>>(qd, kd, vTd, attn);

  // 4) output projection (f32 out + bias)
  gemm_bf16_wmma<2><<<ggrid, gblk, 0, stream>>>(attn, woT, bo, d_out);
}
